// GNN_87960930222500
// MI455X (gfx1250) — matrix operand order repaired and checked by reading, not compiled, not run on hardware
//
#include <hip/hip_runtime.h>

#define IN_F   256
#define HF     256          // HEADS * OUT_F
#define HEADS  4
#define OUT_F  64
#define NEG_SLOPE 0.2f
#define EPSV   1e-16f

// Flip to 0 if the toolchain rejects the async-LDS inline asm.
#ifndef USE_ASYNC_LDS_ASM
#define USE_ASYNC_LDS_ASM 1
#endif

typedef __attribute__((ext_vector_type(16))) __bf16 bf16x16;
typedef __attribute__((ext_vector_type(8)))  float  f32x8;

// ---------------------------------------------------------------------------
// Pack x (f32) -> bf16 WMMA A-fragments.
// A 16x32 bf16 order used here (contiguous k): lane lo = row M; lane hi=0 holds K {0..15},
// hi=1 holds K {16..31}. Fragment elem i: k = hi*16 + i.
// packedA[((tile*8 + kb)*32 + lane)*16 + i]
// blockIdx.x = tile, threadIdx.x = kb*32 + lane.
// ---------------------------------------------------------------------------
__global__ __launch_bounds__(256) void pack_x_frag(
    const float* __restrict__ x, unsigned short* __restrict__ pA, int n_nodes)
{
  const int tid  = threadIdx.x;
  const int lane = tid & 31;
  const int kb   = tid >> 5;          // 0..7
  const int tile = blockIdx.x;
  const int hi   = lane >> 4, lo = lane & 15;

  int row = tile * 16 + lo;
  if (row >= n_nodes) row = n_nodes - 1;      // tail replicate (never stored)
  const float* ap = x + (size_t)row * IN_F + kb * 32 + hi * 16;
  float4 p0 = *(const float4*)(ap + 0);
  float4 p1 = *(const float4*)(ap + 4);
  float4 q0 = *(const float4*)(ap + 8);
  float4 q1 = *(const float4*)(ap + 12);
  float av[16] = {p0.x,p0.y,p0.z,p0.w, p1.x,p1.y,p1.z,p1.w,
                  q0.x,q0.y,q0.z,q0.w, q1.x,q1.y,q1.z,q1.w};
  bf16x16 f;
#pragma unroll
  for (int i = 0; i < 16; ++i) f[i] = (__bf16)av[i];
  ((bf16x16*)pA)[(size_t)tile * 256 + tid] = f;   // 32B contiguous per thread
}

// ---------------------------------------------------------------------------
// Pack W (f32) -> bf16 WMMA B-fragments.
// B 32x16 bf16 order used here (contiguous k, as A): lane lo = col N; hi selects K 0..15 vs 16..31.
// Fragment elem i: k = kb*32 + hi*16 + i.
// packedB[((nt*8 + kb)*32 + lane)*16 + i]; blockIdx.x = nt (0..15).
// ---------------------------------------------------------------------------
__global__ __launch_bounds__(256) void pack_w_frag(
    const float* __restrict__ W, unsigned short* __restrict__ pB)
{
  const int tid  = threadIdx.x;
  const int lane = tid & 31;
  const int kb   = tid >> 5;
  const int nt   = blockIdx.x;
  const int hi   = lane >> 4, lo = lane & 15;

  const int col = nt * 16 + lo;
  const float* wp = W + (size_t)(kb * 32 + hi * 16) * HF + col;
  bf16x16 f;
#pragma unroll
  for (int i = 0; i < 16; ++i) f[i] = (__bf16)wp[(size_t)i * HF];
  ((bf16x16*)pB)[(size_t)nt * 256 + tid] = f;
}

// ---------------------------------------------------------------------------
// GEMM: h = x @ W. Block = 8 waves; wave w owns row-tile (blockIdx.x*8 + w)
// and computes its full 16x256 output band (16 accumulators).
// Packed B (128KB) is staged through LDS in two 64KB phases via
// global_load_async_to_lds_b128 (ASYNCcnt).
// ---------------------------------------------------------------------------
__global__ __launch_bounds__(256) void gat_gemm_wmma(
    const unsigned short* __restrict__ pA, const unsigned short* __restrict__ pB,
    float* __restrict__ h, int n_nodes, int ntiles)
{
  __shared__ uint4 sB[4096];                 // 64KB: 8 n-tiles of B fragments

  const int tid  = threadIdx.x;
  const int wave = tid >> 5;
  const int lane = tid & 31;
  const int hi   = lane >> 4, lo = lane & 15;
  const int mtile  = blockIdx.x * 8 + wave;
  const bool active = (mtile < ntiles);
  const int m0 = mtile * 16;

  const bf16x16* Abase = (const bf16x16*)pA;   // frag idx = (mtile*8+kb)*32+lane
  const uint4*   gB    = (const uint4*)pB;     // 8192 uint4 total

  f32x8 zero = {};
  f32x8 acc[16];
#pragma unroll
  for (int i = 0; i < 16; ++i) acc[i] = zero;

#pragma unroll
  for (int ph = 0; ph < 2; ++ph) {
    // ---- stage 64KB of packed B into LDS --------------------------------
    const uint4* src = gB + ph * 4096;
#if USE_ASYNC_LDS_ASM
#pragma unroll
    for (int it = 0; it < 16; ++it) {
      int idx = it * 256 + tid;                // per-lane 16B chunk
      unsigned lds_off = (unsigned)idx * 16u;  // sB is the only LDS object -> base 0
      // NOTE: dummy "v"(sB) operand makes sB escape so the "memory" clobber
      // aliases it -- otherwise the compiler proves the asm can't write sB
      // and folds all the ds_load reads to undef.
      asm volatile("global_load_async_to_lds_b128 %0, %1, off"
                   :: "v"(lds_off), "v"(src + idx), "v"((const void*)sB)
                   : "memory");
    }
    asm volatile("s_wait_asynccnt 0x0" ::: "memory");
#else
#pragma unroll
    for (int it = 0; it < 16; ++it) {
      int idx = it * 256 + tid;
      sB[idx] = src[idx];
    }
#endif
    __syncthreads();

    // ---- compute: 8 n-tiles against this wave's 16-row band -------------
    if (active) {
      for (int kb = 0; kb < 8; ++kb) {
        bf16x16 a = Abase[(size_t)(mtile * 8 + kb) * 32 + lane];
        if (kb < 7)
          __builtin_prefetch(Abase + (size_t)(mtile * 8 + kb + 1) * 32 + lane, 0, 3);
#pragma unroll
        for (int nt = 0; nt < 8; ++nt) {
          const bf16x16 b = *(const bf16x16*)&sB[((nt * 8 + kb) * 32 + lane) * 2];
          acc[ph * 8 + nt] = __builtin_amdgcn_wmma_f32_16x16x32_bf16(
              false, a, false, b, (short)0, acc[ph * 8 + nt], false, false);
        }
      }
    }
    __syncthreads();   // safe to restage LDS
  }

  // ---- store: C/D 16x16 f32: VGPR r -> row m0 + r + 8*hi, col nt*16 + lo --
  if (!active) return;
  if (m0 + 16 <= n_nodes) {
#pragma unroll
    for (int nt = 0; nt < 16; ++nt)
#pragma unroll
      for (int r = 0; r < 8; ++r)
        h[(size_t)(m0 + r + 8 * hi) * HF + nt * 16 + lo] = acc[nt][r];
  } else {
#pragma unroll
    for (int nt = 0; nt < 16; ++nt)
#pragma unroll
      for (int r = 0; r < 8; ++r) {
        int m = m0 + r + 8 * hi;
        if (m < n_nodes) h[(size_t)m * HF + nt * 16 + lo] = acc[nt][r];
      }
  }
}

// ---------------------------------------------------------------------------
// Per-(node,head) attention dots; also init e_max / e_sum.
// ---------------------------------------------------------------------------
__global__ __launch_bounds__(256) void att_dots(
    const float* __restrict__ h,
    const float* __restrict__ att_src, const float* __restrict__ att_dst,
    float* __restrict__ a_src, float* __restrict__ a_dst,
    float* __restrict__ e_max, float* __restrict__ e_sum, int n_nodes)
{
  int wid  = (blockIdx.x * blockDim.x + threadIdx.x) >> 5;
  int lane = threadIdx.x & 31;
  if (wid >= n_nodes * HEADS) return;
  int node = wid >> 2, hd = wid & 3;

  const float* row = h + (size_t)node * HF + hd * OUT_F;
  const float* as  = att_src + hd * OUT_F;
  const float* ad  = att_dst + hd * OUT_F;
  float s1 = row[lane] * as[lane] + row[lane + 32] * as[lane + 32];
  float s2 = row[lane] * ad[lane] + row[lane + 32] * ad[lane + 32];
#pragma unroll
  for (int off = 16; off; off >>= 1) {
    s1 += __shfl_down(s1, off, 32);
    s2 += __shfl_down(s2, off, 32);
  }
  if (lane == 0) {
    a_src[wid] = s1;
    a_dst[wid] = s2;
    e_max[wid] = -1e30f;
    e_sum[wid] = 0.0f;
  }
}

__global__ void zero_f32(float* __restrict__ p, int total)
{
  int i = blockIdx.x * blockDim.x + threadIdx.x;
  if (i < total) p[i] = 0.0f;
}

// ---------------------------------------------------------------------------
// Edge logits (leaky relu) + segment max via float atomicMax.
// ---------------------------------------------------------------------------
__global__ __launch_bounds__(256) void edge_logits(
    const long long* __restrict__ ei, int n_edges, int n_nodes,
    const float* __restrict__ a_src, const float* __restrict__ a_dst,
    float* __restrict__ e_buf, float* __restrict__ e_max)
{
  int eid = blockIdx.x * blockDim.x + threadIdx.x;
  int total = n_edges + n_nodes;
  if (eid >= total) return;
  int s, d;
  if (eid < n_edges) { s = (int)ei[eid]; d = (int)ei[n_edges + eid]; }
  else               { s = d = eid - n_edges; }
#pragma unroll
  for (int hd = 0; hd < HEADS; ++hd) {
    float e = a_src[s * HEADS + hd] + a_dst[d * HEADS + hd];
    e = (e > 0.0f) ? e : NEG_SLOPE * e;
    e_buf[(size_t)eid * HEADS + hd] = e;
    atomicMax(&e_max[d * HEADS + hd], e);
  }
}

// ---------------------------------------------------------------------------
// exp(e - max) (stored back over e_buf) + segment sum.
// ---------------------------------------------------------------------------
__global__ __launch_bounds__(256) void edge_expsum(
    const long long* __restrict__ ei, int n_edges, int n_nodes,
    float* __restrict__ e_buf, const float* __restrict__ e_max,
    float* __restrict__ e_sum)
{
  int eid = blockIdx.x * blockDim.x + threadIdx.x;
  int total = n_edges + n_nodes;
  if (eid >= total) return;
  int d = (eid < n_edges) ? (int)ei[n_edges + eid] : (eid - n_edges);
#pragma unroll
  for (int hd = 0; hd < HEADS; ++hd) {
    float w = __expf(e_buf[(size_t)eid * HEADS + hd] - e_max[d * HEADS + hd]);
    e_buf[(size_t)eid * HEADS + hd] = w;
    atomicAdd(&e_sum[d * HEADS + hd], w);
  }
}

// ---------------------------------------------------------------------------
// Normalized scatter: out[d] += alpha * h[src]. One wave per edge; h/out are
// L2-resident (51MB each vs 192MB L2).
// ---------------------------------------------------------------------------
__global__ __launch_bounds__(256) void edge_scatter(
    const long long* __restrict__ ei, int n_edges, int n_nodes,
    const float* __restrict__ e_buf, const float* __restrict__ e_sum,
    const float* __restrict__ h, float* __restrict__ out)
{
  long long tid = (long long)blockIdx.x * blockDim.x + threadIdx.x;
  int wid  = (int)(tid >> 5);
  int lane = threadIdx.x & 31;
  int total = n_edges + n_nodes;
  if (wid >= total) return;
  int s, d;
  if (wid < n_edges) { s = (int)ei[wid]; d = (int)ei[n_edges + wid]; }
  else               { s = d = wid - n_edges; }

  const float* hrow = h   + (size_t)s * HF;
  float*       orow = out + (size_t)d * HF;
#pragma unroll
  for (int hd = 0; hd < HEADS; ++hd) {
    float alpha = e_buf[(size_t)wid * HEADS + hd]
                / (e_sum[d * HEADS + hd] + EPSV);
    int f = hd * OUT_F + lane;
    atomicAdd(&orow[f],      alpha * hrow[f]);
    atomicAdd(&orow[f + 32], alpha * hrow[f + 32]);
  }
}

// ---------------------------------------------------------------------------
// out = relu(out + bias)
// ---------------------------------------------------------------------------
__global__ void gat_finish(float* __restrict__ out,
                           const float* __restrict__ bias, int total)
{
  int i = blockIdx.x * blockDim.x + threadIdx.x;
  if (i < total) {
    float v = out[i] + bias[i & (HF - 1)];
    out[i] = (v > 0.0f) ? v : 0.0f;
  }
}

// ---------------------------------------------------------------------------
extern "C" void kernel_launch(void* const* d_in, const int* in_sizes, int n_in,
                              void* d_out, int out_size, void* d_ws, size_t ws_size,
                              hipStream_t stream)
{
  const float*     x       = (const float*)d_in[0];
  const float*     W       = (const float*)d_in[1];
  const float*     att_src = (const float*)d_in[2];
  const float*     att_dst = (const float*)d_in[3];
  const float*     bias    = (const float*)d_in[4];
  const long long* ei      = (const long long*)d_in[5];   // int64 [2, E]

  const int n_nodes = in_sizes[0] / IN_F;
  const int n_edges = in_sizes[5] / 2;
  const int total_e = n_edges + n_nodes;
  const int ntiles  = (n_nodes + 15) / 16;

  // ---- workspace layout -------------------------------------------------
  char* p = (char*)d_ws;
  unsigned short* packA = (unsigned short*)p;            // ntiles*256*16 bf16
  p += (size_t)ntiles * 256 * 16 * 2;
  unsigned short* packB = (unsigned short*)p;            // 16*256*16 bf16 = 128KB
  p += (size_t)16 * 256 * 16 * 2;
  p = (char*)(((uintptr_t)p + 255) & ~(uintptr_t)255);
  float* hbuf  = (float*)p;  p += (size_t)n_nodes * HF * 4;
  float* a_src = (float*)p;  p += (size_t)n_nodes * HEADS * 4;
  float* a_dst = (float*)p;  p += (size_t)n_nodes * HEADS * 4;
  float* e_max = (float*)p;  p += (size_t)n_nodes * HEADS * 4;
  float* e_sum = (float*)p;  p += (size_t)n_nodes * HEADS * 4;
  float* e_buf = (float*)p;                              // total_e * 4 floats
  float* out   = (float*)d_out;

  // ---- pipeline ----------------------------------------------------------
  pack_x_frag<<<ntiles, 256, 0, stream>>>(x, packA, n_nodes);
  pack_w_frag<<<16, 256, 0, stream>>>(W, packB);

  gat_gemm_wmma<<<(ntiles + 7) / 8, 256, 0, stream>>>(packA, packB, hbuf,
                                                      n_nodes, ntiles);

  long long dot_threads = (long long)n_nodes * HEADS * 32;
  att_dots<<<(unsigned)((dot_threads + 255) / 256), 256, 0, stream>>>(
      hbuf, att_src, att_dst, a_src, a_dst, e_max, e_sum, n_nodes);

  zero_f32<<<(out_size + 255) / 256, 256, 0, stream>>>(out, out_size);

  edge_logits<<<(total_e + 255) / 256, 256, 0, stream>>>(
      ei, n_edges, n_nodes, a_src, a_dst, e_buf, e_max);

  edge_expsum<<<(total_e + 255) / 256, 256, 0, stream>>>(
      ei, n_edges, n_nodes, e_buf, e_max, e_sum);

  long long sc_threads = (long long)total_e * 32;
  edge_scatter<<<(unsigned)((sc_threads + 255) / 256), 256, 0, stream>>>(
      ei, n_edges, n_nodes, e_buf, e_sum, hbuf, out);

  gat_finish<<<(out_size + 255) / 256, 256, 0, stream>>>(out, bias, out_size);
}
